// TCLLoss_68925635166275
// MI455X (gfx1250) — matrix operand order repaired and checked by reading, not compiled, not run on hardware
//
#include <hip/hip_runtime.h>

typedef _Float16 v4h  __attribute__((ext_vector_type(4)));
typedef _Float16 v8h  __attribute__((ext_vector_type(8)));
typedef _Float16 v16h __attribute__((ext_vector_type(16)));
typedef float    v8f  __attribute__((ext_vector_type(8)));

#define BT_   256
#define D_    768
#define NP_   196
#define GRID_ 14
#define TFR_  8
#define NPAIR_ 28
#define NPAD_ 224          // 14 col tiles of 16
#define RBLK_ 128          // rows per workgroup (8 row tiles)
#define LSTR_ 40           // halfs per LDS row (80B: conflict-free stride)
#define KC_   32
#define NKC_  24           // 768 / 32
#define NEGF_ (-1.0e30f)
#define LOGEPS_ (-18.420680743952367f)
#define GTEMP_INV (1.0f / 0.07f)
#define PTEMP_INV 10.0f

// ---------------- block reduction helper ----------------
__device__ __forceinline__ float block_reduce256(float v, float* buf, int tid, bool domax) {
    buf[tid] = v;
    __syncthreads();
    for (int s = 128; s > 0; s >>= 1) {
        if (tid < s) buf[tid] = domax ? fmaxf(buf[tid], buf[tid + s]) : (buf[tid] + buf[tid + s]);
        __syncthreads();
    }
    float r = buf[0];
    __syncthreads();
    return r;
}

// ---------------- kernel 1: normalize frame features (fp32) ----------------
__global__ __launch_bounds__(256) void tcl_norm_frames(const float* __restrict__ ff,
                                                       float* __restrict__ featsN) {
    __shared__ float buf[256];
    const int r = blockIdx.x;
    const int tid = threadIdx.x;
    float ss = 0.0f;
    for (int k = tid; k < D_; k += 256) { float v = ff[(size_t)r * D_ + k]; ss += v * v; }
    float tot = block_reduce256(ss, buf, tid, false);
    float inv = 1.0f / fmaxf(sqrtf(tot), 1e-8f);
    for (int k = tid; k < D_; k += 256) featsN[(size_t)r * D_ + k] = ff[(size_t)r * D_ + k] * inv;
}

// ---------------- kernel 2: per-row inverse norms for patches ----------------
__global__ __launch_bounds__(256) void tcl_patch_norms(const float* __restrict__ sp,
                                                       const float* __restrict__ tp,
                                                       float* __restrict__ inv_s,
                                                       float* __restrict__ inv_t) {
    const int row = blockIdx.x * 8 + (threadIdx.x >> 5);   // 0 .. 50175
    const int lane = threadIdx.x & 31;
    const float* ps = sp + (size_t)row * D_;
    const float* pt = tp + (size_t)row * D_;
    float ss = 0.0f, st = 0.0f;
    for (int k = lane; k < D_; k += 32) {
        float a = ps[k]; ss += a * a;
        float b = pt[k]; st += b * b;
    }
    #pragma unroll
    for (int off = 1; off < 32; off <<= 1) {
        ss += __shfl_xor(ss, off);
        st += __shfl_xor(st, off);
    }
    if (lane == 0) {
        inv_s[row] = 1.0f / fmaxf(sqrtf(ss), 1e-12f);
        inv_t[row] = 1.0f / fmaxf(sqrtf(st), 1e-12f);
    }
}

// ---------------- kernel 3: logZ table for the spatial prior ----------------
__global__ __launch_bounds__(256) void tcl_logz(float* __restrict__ logZ) {
    const int idx = blockIdx.x * 256 + threadIdx.x;
    if (idx >= 7 * NP_) return;
    const int dt = idx / NP_ + 1;
    const int n  = idx % NP_;
    const float sigma = 2.0f + 0.5f * (float)dt;
    const float inv2s2 = 1.0f / (2.0f * sigma * sigma);
    const int ni = n / GRID_, nj = n % GRID_;
    float sum = 0.0f;
    for (int m = 0; m < NP_; ++m) {
        const float dx = (float)(ni - m / GRID_);
        const float dy = (float)(nj - m % GRID_);
        sum += __expf(-(dx * dx + dy * dy) * inv2s2);
    }
    logZ[idx] = __logf(sum);
}

// ---------------- kernel 4: global InfoNCE loss (per-row partial) ----------------
__global__ __launch_bounds__(256) void tcl_global_loss(const float* __restrict__ featsN,
                                                       float* __restrict__ gpart) {
    __shared__ float rowi[D_];
    __shared__ float buf[256];
    const int i = blockIdx.x;
    const int tid = threadIdx.x;
    for (int k = tid; k < D_; k += 256) rowi[k] = featsN[(size_t)i * D_ + k];
    __syncthreads();
    const int j = tid;
    const float* rj = featsN + (size_t)j * D_;
    float dot = 0.0f;
    for (int k = 0; k < D_; k += 4) {
        dot += rowi[k] * rj[k] + rowi[k + 1] * rj[k + 1]
             + rowi[k + 2] * rj[k + 2] + rowi[k + 3] * rj[k + 3];
    }
    const float s = fminf(fmaxf(dot * GTEMP_INV, -100.0f), 100.0f);
    const bool self = (j == i);
    const bool neg = !self;
    const bool pos = (!self) && ((j >> 3) == (i >> 3));   // same sequence (T=8)
    const float Mn = block_reduce256(neg ? s : -3.0e38f, buf, tid, true);
    const float Sn = block_reduce256(neg ? __expf(s - Mn) : 0.0f, buf, tid, false);
    const float Mp = block_reduce256(pos ? s : -3.0e38f, buf, tid, true);
    const float Sp = block_reduce256(pos ? __expf(s - Mp) : 0.0f, buf, tid, false);
    if (tid == 0) gpart[i] = (Mn + __logf(Sn)) - (Mp + __logf(Sp));
}

// ---------------- LDS B-fragment loader (32x16 f16 layout) ----------------
__device__ __forceinline__ v16h load_frag16(const _Float16* base, int row, int off) {
    v8h lo = *(const v8h*)(base + row * LSTR_ + off);
    v8h hi = *(const v8h*)(base + row * LSTR_ + off + 16);
    return __builtin_shufflevector(lo, hi, 0,1,2,3,4,5,6,7,8,9,10,11,12,13,14,15);
}

// ---------------- kernel 5: patch loss (WMMA f16) ----------------
__global__ __launch_bounds__(512) void tcl_patch_kernel(
    const float* __restrict__ sp, const float* __restrict__ tp,
    const float* __restrict__ inv_s, const float* __restrict__ inv_t,
    const float* __restrict__ logZ, float* __restrict__ ppart)
{
    // staging buffer: A (2x128 rows) + B (2x224 rows), 40-half stride = 56320 B
    __shared__ __align__(16) _Float16 hbuf[(RBLK_ * 2 + NPAD_ * 2) * LSTR_];
    // post-pass reduction arrays alias the (then-dead) staging buffer
    float (*red)[16][2][6] = (float (*)[16][2][6])hbuf;        // 8*16*2*6 floats
    float* rbuf = ((float*)hbuf) + 8 * 16 * 2 * 6;             // 128 floats

    const int p  = blockIdx.x;   // pair 0..27
    const int b  = blockIdx.y;   // batch 0..31
    const int rb = blockIdx.z;   // 128-row block 0..1
    const int tid  = threadIdx.x;
    const int wave = tid >> 5;
    const int lane = tid & 31;
    const int rt   = wave >> 1;      // row tile within block (0..7)
    const int half = wave & 1;       // column half (0: tiles 0..6, 1: tiles 7..13)
    const int cgrp = lane >> 4;      // hi/lo 16-lane group
    const int l16  = lane & 15;
    const int row0 = rb * RBLK_;
    const bool live = (row0 + rt * 16) < NP_;   // row tile has at least one valid row

    // decode triu pair (ti, tj)
    int ti = 0, rem = p;
    while (rem >= 7 - ti) { rem -= 7 - ti; ++ti; }
    const int tj = ti + 1 + rem;
    const int dt = tj - ti;
    const float sigma = 2.0f + 0.5f * (float)dt;
    const float inv2s2 = 1.0f / (2.0f * sigma * sigma);

    const size_t fi = (size_t)(b * TFR_ + ti);
    const size_t fj = (size_t)(b * TFR_ + tj);
    const float* Ti = tp + fi * (NP_ * D_);
    const float* Tj = tp + fj * (NP_ * D_);
    const float* Si = sp + fi * (NP_ * D_);
    const float* Sj = sp + fj * (NP_ * D_);
    const float* iTi = inv_t + fi * NP_;
    const float* iTj = inv_t + fj * NP_;
    const float* iSi = inv_s + fi * NP_;
    const float* iSj = inv_s + fj * NP_;

    _Float16* A_t = hbuf;
    _Float16* A_s = hbuf + RBLK_ * LSTR_;
    _Float16* B_t = hbuf + 2 * RBLK_ * LSTR_;
    _Float16* B_s = hbuf + (2 * RBLK_ + NPAD_) * LSTR_;

    v8f accT[7], accS[7];
    #pragma unroll
    for (int t = 0; t < 7; ++t) {
        #pragma unroll
        for (int v = 0; v < 8; ++v) { accT[t][v] = 0.0f; accS[t][v] = 0.0f; }
    }

    for (int kc = 0; kc < NKC_; ++kc) {
        const int K0 = kc * KC_;
        __syncthreads();   // previous iteration's fragment reads done

        // ---- stage A tiles: 128 rows x 32 halfs, teacher+student frame ti ----
        for (int idx = tid; idx < RBLK_ * 8; idx += 512) {
            const int r = idx >> 3, q = idx & 7;
            const int n = row0 + r;
            float4 vt = make_float4(0.f, 0.f, 0.f, 0.f), vs = vt;
            float it = 0.0f, is = 0.0f;
            if (n < NP_) {
                vt = *(const float4*)(Ti + (size_t)n * D_ + K0 + q * 4);
                vs = *(const float4*)(Si + (size_t)n * D_ + K0 + q * 4);
                it = iTi[n]; is = iSi[n];
            }
            v4h ht, hs;
            ht[0] = (_Float16)(vt.x * it); ht[1] = (_Float16)(vt.y * it);
            ht[2] = (_Float16)(vt.z * it); ht[3] = (_Float16)(vt.w * it);
            hs[0] = (_Float16)(vs.x * is); hs[1] = (_Float16)(vs.y * is);
            hs[2] = (_Float16)(vs.z * is); hs[3] = (_Float16)(vs.w * is);
            *(v4h*)(A_t + r * LSTR_ + q * 4) = ht;
            *(v4h*)(A_s + r * LSTR_ + q * 4) = hs;
        }
        // ---- stage B tiles: 224 rows x 32 halfs, teacher+student frame tj ----
        for (int idx = tid; idx < NPAD_ * 8; idx += 512) {
            const int m = idx >> 3, q = idx & 7;
            float4 vt = make_float4(0.f, 0.f, 0.f, 0.f), vs = vt;
            float it = 0.0f, is = 0.0f;
            if (m < NP_) {
                vt = *(const float4*)(Tj + (size_t)m * D_ + K0 + q * 4);
                vs = *(const float4*)(Sj + (size_t)m * D_ + K0 + q * 4);
                it = iTj[m]; is = iSj[m];
            }
            v4h ht, hs;
            ht[0] = (_Float16)(vt.x * it); ht[1] = (_Float16)(vt.y * it);
            ht[2] = (_Float16)(vt.z * it); ht[3] = (_Float16)(vt.w * it);
            hs[0] = (_Float16)(vs.x * is); hs[1] = (_Float16)(vs.y * is);
            hs[2] = (_Float16)(vs.z * is); hs[3] = (_Float16)(vs.w * is);
            *(v4h*)(B_t + m * LSTR_ + q * 4) = ht;
            *(v4h*)(B_s + m * LSTR_ + q * 4) = hs;
        }
        __syncthreads();

        // ---- prefetch next K-chunk (one lane per 128B-aligned row chunk) ----
        if ((tid & 7) == 0 && (kc + 1) < NKC_) {
            const int K1 = K0 + KC_;
            for (int r = (tid >> 3); r < RBLK_; r += 64) {
                const int n = row0 + r;
                if (n < NP_) {
                    __builtin_prefetch(Ti + (size_t)n * D_ + K1, 0, 1);
                    __builtin_prefetch(Si + (size_t)n * D_ + K1, 0, 1);
                }
            }
            for (int m = (tid >> 3); m < NP_; m += 64) {
                __builtin_prefetch(Tj + (size_t)m * D_ + K1, 0, 1);
                __builtin_prefetch(Sj + (size_t)m * D_ + K1, 0, 1);
            }
        }

        if (live) {
            // A fragments (16x32 f16: lanes<16 K{0-7,16-23}, lanes>=16 K{8-15,24-31})
            const int arow = rt * 16 + l16;
            v8h alo_t = *(const v8h*)(A_t + arow * LSTR_ + 8 * cgrp);
            v8h ahi_t = *(const v8h*)(A_t + arow * LSTR_ + 16 + 8 * cgrp);
            v8h alo_s = *(const v8h*)(A_s + arow * LSTR_ + 8 * cgrp);
            v8h ahi_s = *(const v8h*)(A_s + arow * LSTR_ + 16 + 8 * cgrp);
            v16h aT = __builtin_shufflevector(alo_t, ahi_t, 0,1,2,3,4,5,6,7,8,9,10,11,12,13,14,15);
            v16h aS = __builtin_shufflevector(alo_s, ahi_s, 0,1,2,3,4,5,6,7,8,9,10,11,12,13,14,15);

            // B fragments, software-pipelined (rotate 2 register sets)
            const int brow0 = (half * 7) * 16 + l16;
            v16h bT_cur = load_frag16(B_t, brow0, 8 * cgrp);
            v16h bS_cur = load_frag16(B_s, brow0, 8 * cgrp);
            #pragma unroll
            for (int t = 0; t < 7; ++t) {
                v16h bT_nxt = bT_cur, bS_nxt = bS_cur;
                if (t < 6) {
                    const int brow = (half * 7 + t + 1) * 16 + l16;
                    bT_nxt = load_frag16(B_t, brow, 8 * cgrp);
                    bS_nxt = load_frag16(B_s, brow, 8 * cgrp);
                }
                accT[t] = __builtin_amdgcn_wmma_f32_16x16x32_f16(false, aT, false, bT_cur,
                                                                 (short)0, accT[t], false, false);
                accS[t] = __builtin_amdgcn_wmma_f32_16x16x32_f16(false, aS, false, bS_cur,
                                                                 (short)0, accS[t], false, false);
                bT_cur = bT_nxt; bS_cur = bS_nxt;
            }
        }
    }

    __syncthreads();   // all hbuf reads done; red/rbuf alias becomes safe

    // ---- post-process: CE_row = lse_s - W/St, from register tiles ----
    if (live) {
        #pragma unroll
        for (int v = 0; v < 8; ++v) {
            const int nloc = rt * 16 + v + 8 * cgrp;
            const int n = row0 + nloc;
            const bool nv = (n < NP_);
            const int ni = nv ? (n / GRID_) : 0;
            const int nj = nv ? (n % GRID_) : 0;
            const float lz = nv ? logZ[(dt - 1) * NP_ + n] : 0.0f;
            float mt = NEGF_, ms = NEGF_;
            #pragma unroll
            for (int t = 0; t < 7; ++t) {
                const int m = (half * 7 + t) * 16 + l16;
                float lt = NEGF_, ls = NEGF_;
                if (nv && m < NP_) {
                    const float te = fminf(fmaxf(accT[t][v] * PTEMP_INV, -100.0f), 100.0f);
                    const float dx = (float)(ni - m / GRID_);
                    const float dy = (float)(nj - m % GRID_);
                    const float lp = fmaxf(-(dx * dx + dy * dy) * inv2s2 - lz, LOGEPS_);
                    lt = te + lp;
                    ls = fminf(fmaxf(accS[t][v] * PTEMP_INV, -100.0f), 100.0f);
                }
                accT[t][v] = lt;   // overwrite with final logits
                accS[t][v] = ls;
                mt = fmaxf(mt, lt);
                ms = fmaxf(ms, ls);
            }
            #pragma unroll
            for (int off = 1; off < 16; off <<= 1) {   // reduce within 16-lane row group
                mt = fmaxf(mt, __shfl_xor(mt, off));
                ms = fmaxf(ms, __shfl_xor(ms, off));
            }
            float st = 0.0f, ss = 0.0f, w = 0.0f;
            #pragma unroll
            for (int t = 0; t < 7; ++t) {
                const float lt = accT[t][v], ls = accS[t][v];
                const float et = __expf(lt - mt);     // 0 for invalid entries
                st += et;
                w  += et * ls;                        // et==0 kills invalid ls (finite NEGF_)
                ss += __expf(ls - ms);
            }
            #pragma unroll
            for (int off = 1; off < 16; off <<= 1) {
                st += __shfl_xor(st, off);
                ss += __shfl_xor(ss, off);
                w  += __shfl_xor(w,  off);
            }
            if (l16 == 0) {
                float* q = &red[rt][v + 8 * cgrp][half][0];
                q[0] = mt; q[1] = st; q[2] = w; q[3] = ms; q[4] = ss;
            }
        }
    }
    __syncthreads();

    float ce = 0.0f;
    if (tid < RBLK_) {
        const int n = row0 + tid;
        if (n < NP_) {
            const int rt2 = tid >> 4, r2 = tid & 15;
            const float mt0 = red[rt2][r2][0][0], st0 = red[rt2][r2][0][1], w0 = red[rt2][r2][0][2];
            const float ms0 = red[rt2][r2][0][3], ss0 = red[rt2][r2][0][4];
            const float mt1 = red[rt2][r2][1][0], st1 = red[rt2][r2][1][1], w1 = red[rt2][r2][1][2];
            const float ms1 = red[rt2][r2][1][3], ss1 = red[rt2][r2][1][4];
            const float mtx = fmaxf(mt0, mt1);
            const float e0 = __expf(mt0 - mtx), e1 = __expf(mt1 - mtx);
            const float st = st0 * e0 + st1 * e1;
            const float w  = w0 * e0 + w1 * e1;
            const float msx = fmaxf(ms0, ms1);
            const float ss = ss0 * __expf(ms0 - msx) + ss1 * __expf(ms1 - msx);
            ce = (msx + __logf(ss)) - w / st;
        }
        rbuf[tid] = ce;
    }
    __syncthreads();
    if (tid == 0) {
        float s = 0.0f;
        for (int k = 0; k < RBLK_; ++k) s += rbuf[k];
        ppart[(b * NPAIR_ + p) * 2 + rb] = s;
    }
}

// ---------------- kernel 6: deterministic finalize ----------------
__global__ __launch_bounds__(256) void tcl_finalize(const float* __restrict__ gpart,
                                                    const float* __restrict__ ppart,
                                                    float* __restrict__ out) {
    __shared__ float buf[256];
    const int tid = threadIdx.x;
    float gs = block_reduce256(gpart[tid], buf, tid, false);
    if (tid == 0) out[0] = gs / 256.0f;
    float s = 0.0f;
    for (int k = tid; k < 32 * NPAIR_ * 2; k += 256) s += ppart[k];
    float ps = block_reduce256(s, buf, tid, false);
    if (tid == 0) out[1] = ps / (float)(32 * NPAIR_ * NP_);
}

// ---------------- launcher ----------------
extern "C" void kernel_launch(void* const* d_in, const int* in_sizes, int n_in,
                              void* d_out, int out_size, void* d_ws, size_t ws_size,
                              hipStream_t stream) {
    (void)in_sizes; (void)n_in; (void)out_size; (void)ws_size;
    const float* ff = (const float*)d_in[0];   // frame_features (256, 768)
    const float* sp = (const float*)d_in[1];   // student_patches (256, 196, 768)
    const float* tp = (const float*)d_in[2];   // teacher_patches (256, 196, 768)

    float* ws = (float*)d_ws;
    float* featsN = ws;                      // 196608
    float* inv_s  = featsN + 196608;         // 50176
    float* inv_t  = inv_s + 50176;           // 50176
    float* logZ   = inv_t + 50176;           // 1372
    float* gpart  = logZ + 1372;             // 256
    float* ppart  = gpart + 256;             // 1792
    float* out = (float*)d_out;

    tcl_norm_frames<<<256, 256, 0, stream>>>(ff, featsN);
    tcl_patch_norms<<<6272, 256, 0, stream>>>(sp, tp, inv_s, inv_t);
    tcl_logz<<<6, 256, 0, stream>>>(logZ);
    tcl_global_loss<<<256, 256, 0, stream>>>(featsN, gpart);
    tcl_patch_kernel<<<dim3(NPAIR_, 32, 2), 512, 0, stream>>>(sp, tp, inv_s, inv_t, logZ, ppart);
    tcl_finalize<<<1, 256, 0, stream>>>(gpart, ppart, out);
}
